// benchmark_28398323761484
// MI455X (gfx1250) — compile-verified, matrix operand order checked by reading, not run on hardware
//
#include <hip/hip_runtime.h>
#include <hip/hip_bf16.h>
#include <math.h>

typedef __attribute__((ext_vector_type(16))) __bf16 v16bf;
typedef __attribute__((ext_vector_type(8)))  __bf16 v8bf;
typedef __attribute__((ext_vector_type(8)))  float  v8f;

#define LROWS 4096
#define HEADS 8
#define HDIM 32
#define KNEI 10

#define USE_ASYNC_LDS 1

// float -> bf16 round-to-nearest-even
static __device__ __forceinline__ __bf16 f2bf(float f) {
  unsigned u = __float_as_uint(f);
  unsigned r = (u + 0x7fffu + ((u >> 16) & 1u)) >> 16;
  unsigned short h = (unsigned short)r;
  __bf16 b;
  __builtin_memcpy(&b, &h, 2);
  return b;
}

#if USE_ASYNC_LDS
// gfx1250 async copy: global -> LDS, 16 bytes per lane, tracked by ASYNCcnt.
// LDS byte address = low 32 bits of the generic pointer (flat->LDS truncation).
static __device__ __forceinline__ void async_cp_b128(void* lds, const void* g) {
  unsigned l = (unsigned)(unsigned long long)lds;
  unsigned long long ga = (unsigned long long)g;
  asm volatile("global_load_async_to_lds_b128 %0, %1, off"
               :: "v"(l), "v"(ga) : "memory");
}
static __device__ __forceinline__ void async_wait0() {
  asm volatile("s_wait_asynccnt 0" ::: "memory");
}
static __device__ __forceinline__ void async_wait3() {
  asm volatile("s_wait_asynccnt 3" ::: "memory");
}
#endif

// ---------------------------------------------------------------------------
// Conversion kernels (run once per launch; removes cvt VALU from GEMM loops)
// ---------------------------------------------------------------------------
__global__ __launch_bounds__(256)
void cvt_bf16_kernel(const float* __restrict__ s, __bf16* __restrict__ d, int n) {
  int i = blockIdx.x * 256 + threadIdx.x;
  if (i < n) d[i] = f2bf(s[i]);
}

// W[K x N] f32 row-major  ->  Wt[N x K] bf16 row-major (B fragments contiguous)
__global__ __launch_bounds__(256)
void cvt_bf16_transpose_kernel(const float* __restrict__ W, __bf16* __restrict__ Wt,
                               int K, int N) {
  int i = blockIdx.x * 256 + threadIdx.x;
  if (i < K * N) {
    int k = i / N, n = i - k * N;
    Wt[(size_t)n * K + k] = f2bf(W[i]);
  }
}

// ---------------------------------------------------------------------------
// GEMM: C[L x N] = X[L x K] @ W[K x N], X bf16 row-major, Wt bf16 [N x K].
// Block 256 thr = 8 waves; tile 128(M) x 64(N); wave = 16 rows x 4 WMMA
// tiles (A fragment reused 4x). K step 32.  Double-buffered LDS staging via
// global_load_async_to_lds_b128: step s+1's tiles stream in (ASYNCcnt) while
// step s computes; s_wait_asynccnt 3 leaves only the in-flight next-buffer
// loads outstanding.
// ---------------------------------------------------------------------------
__global__ __launch_bounds__(256)
void wmma_gemm_bf16_kernel(const __bf16* __restrict__ X, const __bf16* __restrict__ Wt,
                           float* __restrict__ C, int K, int N) {
  __shared__ __bf16 sA[2][128 * 32];   // 2 x 8 KB : row m-local, col k
  __shared__ __bf16 sB[2][64 * 32];    // 2 x 4 KB : row n-local, col k
  const int tid  = threadIdx.x;
  const int wave = tid >> 5;
  const int lane = tid & 31;
  const int m0 = blockIdx.x * 128;
  const int n0 = blockIdx.y * 64;

  const int half = lane >> 4;       // K sub-stripe select (ISA 16-bit layout)
  const int r    = lane & 15;
  const int kb   = half * 8;

  // staging assignments: A: 32B/thread (2 x b128), B: 16B/thread (1 x b128)
  const int arow  = tid >> 1;            // 0..127
  const int apart = (tid & 1) << 4;      // 0 or 16 elems
  const int brow  = tid >> 2;            // 0..63
  const int bpart = (tid & 3) << 3;      // 0,8,16,24 elems

  v8f acc[4] = {{}, {}, {}, {}};

  auto issue = [&](int buf, int k0) {
    const __bf16* xg = X  + (size_t)(m0 + arow) * K + (k0 + apart);
    const __bf16* wg = Wt + (size_t)(n0 + brow) * K + (k0 + bpart);
#if USE_ASYNC_LDS
    async_cp_b128(&sA[buf][arow * 32 + apart],     xg);
    async_cp_b128(&sA[buf][arow * 32 + apart + 8], xg + 8);
    async_cp_b128(&sB[buf][brow * 32 + bpart],     wg);
#else
    *(uint4*)&sA[buf][arow * 32 + apart]     = *(const uint4*)xg;
    *(uint4*)&sA[buf][arow * 32 + apart + 8] = *(const uint4*)(xg + 8);
    *(uint4*)&sB[buf][brow * 32 + bpart]     = *(const uint4*)wg;
#endif
  };

  const int nsteps = K >> 5;
  issue(0, 0);
  for (int s = 0; s < nsteps; ++s) {
    const int cur = s & 1;
    const bool more = (s + 1) < nsteps;
    if (more) issue(cur ^ 1, (s + 1) << 5);   // prefetch next K slice
#if USE_ASYNC_LDS
    if (more) async_wait3();                  // current buffer's 3 loads done
    else      async_wait0();
#endif
    __syncthreads();                          // all waves' tiles visible

    // A fragment: lanes 0-15 K={0..7,16..23}, lanes 16-31 K={8..15,24..31}
    const __bf16* ap = &sA[cur][(wave * 16 + r) * 32];
    v8bf alo = *(const v8bf*)(ap + kb);
    v8bf ahi = *(const v8bf*)(ap + kb + 16);
    v16bf a = __builtin_shufflevector(alo, ahi, 0, 1, 2, 3, 4, 5, 6, 7,
                                      8, 9, 10, 11, 12, 13, 14, 15);
#pragma unroll
    for (int nt = 0; nt < 4; ++nt) {
      const __bf16* bp = &sB[cur][(nt * 16 + r) * 32];
      v8bf blo = *(const v8bf*)(bp + kb);
      v8bf bhi = *(const v8bf*)(bp + kb + 16);
      v16bf b = __builtin_shufflevector(blo, bhi, 0, 1, 2, 3, 4, 5, 6, 7,
                                        8, 9, 10, 11, 12, 13, 14, 15);
      acc[nt] = __builtin_amdgcn_wmma_f32_16x16x32_bf16(false, a, false, b,
                                                        (short)0, acc[nt],
                                                        false, false);
    }
    __syncthreads();   // protect buffer `cur` from step s+1's issue into it
  }

  // C/D layout: VGPR q holds M = q + 8*half; column = r within each 16-tile
#pragma unroll
  for (int nt = 0; nt < 4; ++nt) {
#pragma unroll
    for (int q = 0; q < 8; ++q) {
      int m = m0 + wave * 16 + q + 8 * half;
      C[(size_t)m * N + (n0 + nt * 16 + r)] = acc[nt][q];
    }
  }
}

// ---------------------------------------------------------------------------
// Row-wise epilogue (one wave per row):
//   v = maybeLeaky( maybeLN( alpha*res + beta*(Craw + bias) ) )
// writes optional f32 copy and optional bf16 copy (feeds next bf16 GEMM).
// ---------------------------------------------------------------------------
__global__ __launch_bounds__(256)
void epilogue_kernel(const float* __restrict__ Craw, const float* __restrict__ bias,
                     const float* __restrict__ res, const float* __restrict__ alpha_p,
                     const float* __restrict__ beta_p, const float* __restrict__ gamma,
                     const float* __restrict__ lnbeta, float* __restrict__ outf,
                     __bf16* __restrict__ outb,
                     int N, int ldo, int col_off, int do_ln, int do_leaky) {
  const int row  = blockIdx.x * 8 + (threadIdx.x >> 5);
  const int lane = threadIdx.x & 31;
  const int per  = N >> 5;                 // 8 or 16
  float alpha = 1.0f, beta = 1.0f;
  if (alpha_p) alpha = *alpha_p;
  if (beta_p)  beta  = *beta_p;

  float vals[16];
  float s = 0.0f;
  for (int t = 0; t < per; ++t) {
    int c = lane + (t << 5);
    float v = Craw[(size_t)row * N + c] + bias[c];
    if (res) v = alpha * res[(size_t)row * N + c] + beta * v;
    vals[t] = v;
    s += v;
  }
  if (do_ln) {
    for (int off = 16; off > 0; off >>= 1) s += __shfl_xor(s, off, 32);
    float mean = s / (float)N;
    float var = 0.0f;
    for (int t = 0; t < per; ++t) { float d = vals[t] - mean; var += d * d; }
    for (int off = 16; off > 0; off >>= 1) var += __shfl_xor(var, off, 32);
    float rstd = rsqrtf(var / (float)N + 1e-5f);
    for (int t = 0; t < per; ++t) {
      int c = lane + (t << 5);
      vals[t] = (vals[t] - mean) * rstd * gamma[c] + lnbeta[c];
    }
  }
  if (do_leaky)
    for (int t = 0; t < per; ++t)
      vals[t] = vals[t] >= 0.0f ? vals[t] : 0.01f * vals[t];

  for (int t = 0; t < per; ++t) {
    int c = lane + (t << 5);
    size_t o = (size_t)row * ldo + col_off + c;
    if (outf) outf[o] = vals[t];
    if (outb) outb[o] = f2bf(vals[t]);
  }
}

// ---------------------------------------------------------------------------
// Exact kNN (k=10, includes self). One block per query row; 10 arg-min passes
// on packed (distbits<<32)|idx  (matches jax top_k tie-breaking on -d2).
// ---------------------------------------------------------------------------
__global__ __launch_bounds__(256)
void knn_kernel(const float* __restrict__ pos, int* __restrict__ nbr) {
  __shared__ float d2s[LROWS];                       // 16 KB
  __shared__ unsigned long long red[256];
  const int i = blockIdx.x;
  const float qx = pos[i * 3 + 0];
  const float qy = pos[i * 3 + 1];
  const float qz = pos[i * 3 + 2];
  for (int j = threadIdx.x; j < LROWS; j += 256) {
    float dx = pos[j * 3 + 0] - qx;
    float dy = pos[j * 3 + 1] - qy;
    float dz = pos[j * 3 + 2] - qz;
    d2s[j] = dx * dx + dy * dy + dz * dz;
  }
  __syncthreads();
  for (int t = 0; t < KNEI; ++t) {
    unsigned long long best = ~0ull;
    for (int j = threadIdx.x; j < LROWS; j += 256) {
      unsigned long long key =
          (((unsigned long long)__float_as_uint(d2s[j])) << 32) | (unsigned)j;
      best = best < key ? best : key;
    }
    red[threadIdx.x] = best;
    __syncthreads();
    for (int sft = 128; sft > 0; sft >>= 1) {
      if (threadIdx.x < sft) {
        unsigned long long o = red[threadIdx.x + sft];
        if (o < red[threadIdx.x]) red[threadIdx.x] = o;
      }
      __syncthreads();
    }
    if (threadIdx.x == 0) {
      int jmin = (int)(red[0] & 0xffffffffu);
      nbr[i * KNEI + t] = jmin;
      d2s[jmin] = __uint_as_float(0x7f800000u);      // +inf: exclude
    }
    __syncthreads();
  }
}

// ---------------------------------------------------------------------------
// Sparse masked attention; emits bf16 (feeds o-proj WMMA GEMM directly).
// ---------------------------------------------------------------------------
__global__ __launch_bounds__(256)
void sparse_attn_kernel(const float* __restrict__ Q, const float* __restrict__ Km,
                        const float* __restrict__ V, const int* __restrict__ nbr,
                        __bf16* __restrict__ Out) {
  __shared__ int   nb[KNEI];
  __shared__ float sc[HEADS][KNEI];
  __shared__ float qrow[256];
  __shared__ float vrows[KNEI][256];                 // 10 KB
  const int i = blockIdx.x;
  if (threadIdx.x < KNEI) nb[threadIdx.x] = nbr[i * KNEI + threadIdx.x];
  qrow[threadIdx.x] = Q[(size_t)i * 256 + threadIdx.x];
  __syncthreads();

  for (int t = threadIdx.x; t < KNEI * 256; t += 256) {
    int j = t >> 8, d = t & 255;
    vrows[j][d] = V[(size_t)nb[j] * 256 + d];
  }
  if (threadIdx.x < HEADS * KNEI) {
    int h = threadIdx.x / KNEI, j = threadIdx.x % KNEI;
    const float* krow = Km + (size_t)nb[j] * 256 + h * HDIM;
    float s = 0.0f;
#pragma unroll
    for (int d = 0; d < HDIM; ++d) s += qrow[h * HDIM + d] * krow[d];
    sc[h][j] = s * 0.17677669529663687f;             // 1/sqrt(32)
  }
  __syncthreads();

  if (threadIdx.x < HEADS) {
    int h = threadIdx.x;
    float mx = -INFINITY;
    for (int j = 0; j < KNEI; ++j) mx = fmaxf(mx, sc[h][j]);
    float se = 0.0f;
    for (int j = 0; j < KNEI; ++j) { float e = __expf(sc[h][j] - mx); sc[h][j] = e; se += e; }
    float inv = 1.0f / se;
    for (int j = 0; j < KNEI; ++j) sc[h][j] *= inv;
  }
  __syncthreads();

  const int d = threadIdx.x;
  const int h = d >> 5;
  float o = 0.0f;
#pragma unroll
  for (int j = 0; j < KNEI; ++j) o += sc[h][j] * vrows[j][d];
  Out[(size_t)i * 256 + d] = f2bf(o);
}

// ---------------------------------------------------------------------------
extern "C" void kernel_launch(void* const* d_in, const int* in_sizes, int n_in,
                              void* d_out, int out_size, void* d_ws, size_t ws_size,
                              hipStream_t stream) {
  (void)in_sizes; (void)n_in; (void)out_size; (void)ws_size;
  const float* geo_feat = (const float*)d_in[0];
  const float* sem_feat = (const float*)d_in[1];
  const float* rsa_feat = (const float*)d_in[2];
  const float* pos      = (const float*)d_in[3];
  const float* geo_w  = (const float*)d_in[4];
  const float* geo_b  = (const float*)d_in[5];
  const float* geo_g  = (const float*)d_in[6];
  const float* geo_be = (const float*)d_in[7];
  const float* sem_w  = (const float*)d_in[8];
  const float* sem_b  = (const float*)d_in[9];
  const float* sem_g  = (const float*)d_in[10];
  const float* sem_be = (const float*)d_in[11];
  const float* rsa_w  = (const float*)d_in[12];
  const float* rsa_b  = (const float*)d_in[13];
  const float* rsa_g  = (const float*)d_in[14];
  const float* rsa_be = (const float*)d_in[15];
  const float* q_w = (const float*)d_in[16];
  const float* q_b = (const float*)d_in[17];
  const float* k_w = (const float*)d_in[18];
  const float* k_b = (const float*)d_in[19];
  const float* v_w = (const float*)d_in[20];
  const float* v_b = (const float*)d_in[21];
  const float* o_w = (const float*)d_in[22];
  const float* o_b = (const float*)d_in[23];
  const float* alpha_geo = (const float*)d_in[24];
  const float* beta_geo  = (const float*)d_in[25];
  const float* alpha_sem = (const float*)d_in[26];
  const float* beta_sem  = (const float*)d_in[27];
  const float* rt_w  = (const float*)d_in[28];
  const float* rt_b  = (const float*)d_in[29];
  const float* rt_g  = (const float*)d_in[30];
  const float* rt_be = (const float*)d_in[31];
  const float* f1_w  = (const float*)d_in[32];
  const float* f1_b  = (const float*)d_in[33];
  const float* f1_g  = (const float*)d_in[34];
  const float* f1_be = (const float*)d_in[35];
  const float* f2_w  = (const float*)d_in[36];
  const float* f2_b  = (const float*)d_in[37];
  const float* f2_g  = (const float*)d_in[38];
  const float* f2_be = (const float*)d_in[39];
  const float* ln1_g = (const float*)d_in[40];
  const float* ln1_b = (const float*)d_in[41];
  const float* ln2_g = (const float*)d_in[42];
  const float* ln2_b = (const float*)d_in[43];

  // ---- workspace layout (byte allocator, 256B aligned chunks)
  char* base = (char*)d_ws;
  size_t woff = 0;
  auto alloc = [&](size_t bytes) -> void* {
    void* p = base + woff;
    woff = (woff + bytes + 255) & ~(size_t)255;
    return p;
  };
  const size_t R = LROWS;
  // f32 buffers
  float* tmp   = (float*)alloc(R * 512 * 4);      // raw GEMM out (max N=512)
  float* geo_p = (float*)alloc(R * 256 * 4);
  float* sem_p = (float*)alloc(R * 256 * 4);
  float* qg = (float*)alloc(R * 256 * 4);
  float* kg = (float*)alloc(R * 256 * 4);
  float* vg = (float*)alloc(R * 256 * 4);
  float* qs = (float*)alloc(R * 256 * 4);
  float* ks = (float*)alloc(R * 256 * 4);
  float* vs = (float*)alloc(R * 256 * 4);
  // bf16 activation buffers
  __bf16* geo_feat_bf = (__bf16*)alloc(R * 1536 * 2);
  __bf16* sem_feat_bf = (__bf16*)alloc(R * 512 * 2);
  __bf16* rsa_feat_bf = (__bf16*)alloc(R * 64 * 2);
  __bf16* geo_p_bf = (__bf16*)alloc(R * 256 * 2);
  __bf16* sem_p_bf = (__bf16*)alloc(R * 256 * 2);
  __bf16* rsa_p_bf = (__bf16*)alloc(R * 256 * 2);
  __bf16* ag_bf    = (__bf16*)alloc(R * 256 * 2);
  __bf16* as_bf    = (__bf16*)alloc(R * 256 * 2);
  __bf16* comb_bf  = (__bf16*)alloc(R * 768 * 2);
  __bf16* h1_bf    = (__bf16*)alloc(R * 512 * 2);
  // bf16 transposed weights [N x K]
  __bf16* geo_wt = (__bf16*)alloc(1536 * 256 * 2);
  __bf16* sem_wt = (__bf16*)alloc(512 * 256 * 2);
  __bf16* rsa_wt = (__bf16*)alloc(64 * 256 * 2);
  __bf16* q_wt   = (__bf16*)alloc(256 * 256 * 2);
  __bf16* k_wt   = (__bf16*)alloc(256 * 256 * 2);
  __bf16* v_wt   = (__bf16*)alloc(256 * 256 * 2);
  __bf16* o_wt   = (__bf16*)alloc(256 * 256 * 2);
  __bf16* rt_wt  = (__bf16*)alloc(256 * 256 * 2);
  __bf16* f1_wt  = (__bf16*)alloc(768 * 512 * 2);
  __bf16* f2_wt  = (__bf16*)alloc(512 * 256 * 2);
  int* nbr = (int*)alloc(R * KNEI * 4);

  dim3 blk(256);
  auto cvt = [&](const float* s, __bf16* d, int n) {
    cvt_bf16_kernel<<<dim3((n + 255) / 256), blk, 0, stream>>>(s, d, n);
  };
  auto cvtT = [&](const float* W, __bf16* Wt, int K, int N) {
    cvt_bf16_transpose_kernel<<<dim3((K * N + 255) / 256), blk, 0, stream>>>(W, Wt, K, N);
  };
  auto gemm = [&](const __bf16* X, const __bf16* Wt, float* C, int K, int N) {
    dim3 grid(LROWS / 128, N / 64);
    wmma_gemm_bf16_kernel<<<grid, blk, 0, stream>>>(X, Wt, C, K, N);
  };
  auto epi = [&](const float* C, const float* b, const float* res,
                 const float* ap, const float* bp, const float* g, const float* lb,
                 float* outf, __bf16* outb, int N, int ldo, int off, int ln, int lk) {
    epilogue_kernel<<<dim3(LROWS / 8), blk, 0, stream>>>(C, b, res, ap, bp, g, lb,
                                                         outf, outb, N, ldo, off, ln, lk);
  };

  // --- one-time conversions (inputs + weights)
  cvt(geo_feat, geo_feat_bf, LROWS * 1536);
  cvt(sem_feat, sem_feat_bf, LROWS * 512);
  cvt(rsa_feat, rsa_feat_bf, LROWS * 64);
  cvtT(geo_w, geo_wt, 1536, 256);
  cvtT(sem_w, sem_wt, 512, 256);
  cvtT(rsa_w, rsa_wt, 64, 256);
  cvtT(q_w, q_wt, 256, 256);
  cvtT(k_w, k_wt, 256, 256);
  cvtT(v_w, v_wt, 256, 256);
  cvtT(o_w, o_wt, 256, 256);
  cvtT(rt_w, rt_wt, 256, 256);
  cvtT(f1_w, f1_wt, 768, 512);
  cvtT(f2_w, f2_wt, 512, 256);

  // --- input projections + LN  (f32 copy for residual/attn, bf16 for GEMMs)
  gemm(geo_feat_bf, geo_wt, tmp, 1536, 256);
  epi(tmp, geo_b, nullptr, nullptr, nullptr, geo_g, geo_be, geo_p, geo_p_bf, 256, 256, 0, 1, 0);
  gemm(sem_feat_bf, sem_wt, tmp, 512, 256);
  epi(tmp, sem_b, nullptr, nullptr, nullptr, sem_g, sem_be, sem_p, sem_p_bf, 256, 256, 0, 1, 0);
  gemm(rsa_feat_bf, rsa_wt, tmp, 64, 256);
  epi(tmp, rsa_b, nullptr, nullptr, nullptr, rsa_g, rsa_be, nullptr, rsa_p_bf, 256, 256, 0, 1, 0);

  // --- kNN indices
  knn_kernel<<<dim3(LROWS), blk, 0, stream>>>(pos, nbr);

  // --- shared q/k/v projections (bias only)
  gemm(geo_p_bf, q_wt, tmp, 256, 256);
  epi(tmp, q_b, nullptr, nullptr, nullptr, nullptr, nullptr, qg, nullptr, 256, 256, 0, 0, 0);
  gemm(geo_p_bf, k_wt, tmp, 256, 256);
  epi(tmp, k_b, nullptr, nullptr, nullptr, nullptr, nullptr, kg, nullptr, 256, 256, 0, 0, 0);
  gemm(geo_p_bf, v_wt, tmp, 256, 256);
  epi(tmp, v_b, nullptr, nullptr, nullptr, nullptr, nullptr, vg, nullptr, 256, 256, 0, 0, 0);
  gemm(sem_p_bf, q_wt, tmp, 256, 256);
  epi(tmp, q_b, nullptr, nullptr, nullptr, nullptr, nullptr, qs, nullptr, 256, 256, 0, 0, 0);
  gemm(sem_p_bf, k_wt, tmp, 256, 256);
  epi(tmp, k_b, nullptr, nullptr, nullptr, nullptr, nullptr, ks, nullptr, 256, 256, 0, 0, 0);
  gemm(sem_p_bf, v_wt, tmp, 256, 256);
  epi(tmp, v_b, nullptr, nullptr, nullptr, nullptr, nullptr, vs, nullptr, 256, 256, 0, 0, 0);

  // --- sparse kNN attention (bf16 outputs feed o-proj GEMM)
  sparse_attn_kernel<<<dim3(LROWS), blk, 0, stream>>>(qg, ks, vs, nbr, ag_bf);
  sparse_attn_kernel<<<dim3(LROWS), blk, 0, stream>>>(qs, kg, vg, nbr, as_bf);

  // --- output proj + gated residual + LN, concatenated into comb_bf [.,768]
  gemm(ag_bf, o_wt, tmp, 256, 256);
  epi(tmp, o_b, geo_p, alpha_geo, beta_geo, ln1_g, ln1_b, nullptr, comb_bf, 256, 768, 0, 1, 0);
  gemm(as_bf, o_wt, tmp, 256, 256);
  epi(tmp, o_b, sem_p, alpha_sem, beta_sem, ln2_g, ln2_b, nullptr, comb_bf, 256, 768, 256, 1, 0);

  // --- rsa transform: Linear + LN + LeakyReLU, into comb cols 512..767
  gemm(rsa_p_bf, rt_wt, tmp, 256, 256);
  epi(tmp, rt_b, nullptr, nullptr, nullptr, rt_g, rt_be, nullptr, comb_bf, 256, 768, 512, 1, 1);

  // --- fusion MLP
  gemm(comb_bf, f1_wt, tmp, 768, 512);
  epi(tmp, f1_b, nullptr, nullptr, nullptr, f1_g, f1_be, nullptr, h1_bf, 512, 512, 0, 1, 1);
  gemm(h1_bf, f2_wt, tmp, 512, 256);
  epi(tmp, f2_b, nullptr, nullptr, nullptr, f2_g, f2_be, (float*)d_out, nullptr, 256, 256, 0, 1, 1);
}
